// R_Net_40836549050368
// MI455X (gfx1250) — compile-verified, matrix operand order checked by reading, not run on hardware
//
#include <hip/hip_runtime.h>
#include <hip/hip_bf16.h>

typedef __bf16 bf16;
typedef __attribute__((ext_vector_type(16))) __bf16 vbf;
typedef __attribute__((ext_vector_type(8)))  float v8f;

// ---------------------------------------------------------------------------
// Model dims
// ---------------------------------------------------------------------------
#define BSZ   16
#define PL    256
#define QL    64
#define HH    75
#define RR    450
#define EMBD  300
#define PB    (PL*BSZ)   // 4096 rows of (t,b)
#define QB    (QL*BSZ)   // 1024 rows

__device__ __forceinline__ float sigm(float x) { return 1.0f / (1.0f + __expf(-x)); }

// ---------------------------------------------------------------------------
// WMMA tile loaders, following CDNA5 16-bit A/B VGPR layouts (ISA 7.12.2):
// A (16x32, row-major lda): lane m=lane&15 holds row m; elems 0..7 = K[kh..kh+7],
// elems 8..15 = K[16+kh..16+kh+7], kh = (lane>=16)?8:0.
// B (32x16): lane n holds column n == row n of the (N,K) row-major weight,
// identical K packing. C/D f32: lane = N col (lane&15), VGPR v -> M = v + 8*(lane>>4).
// All weight buffers are packed with BOTH K padded to mult-of-32 and N (rows)
// padded to mult-of-16, so tile loads need no guards (no EXEC divergence in
// the WMMA K-loop); only stores are guarded by the true N.
// ---------------------------------------------------------------------------
__device__ __forceinline__ vbf load_tileA(const bf16* A, int lda, int k, int lane) {
    const int m  = lane & 15;
    const int kh = (lane >> 4) << 3;
    const bf16* p = A + (size_t)m * lda + k + kh;
    vbf r;
#pragma unroll
    for (int i = 0; i < 8; ++i) { r[i] = p[i]; r[i + 8] = p[i + 16]; }
    return r;
}

__device__ __forceinline__ vbf load_tileB(const bf16* W, int ldw, int nbase, int k, int lane) {
    const int n  = nbase + (lane & 15);
    const int kh = (lane >> 4) << 3;
    const bf16* p = W + (size_t)n * ldw + k + kh;
    vbf r;
#pragma unroll
    for (int i = 0; i < 8; ++i) { r[i] = p[i]; r[i + 8] = p[i + 16]; }
    return r;
}

// Block-level GEMM: OUT[16 x N] (f32, ldo) (+)= A[16 x Kpad bf16, lda] @ W[Npad x Kpad]^T (+bias)
// One 16x16 N-tile per wave per iteration; uniform control flow per wave (EXEC all 1s).
template <bool ACCUM>
__device__ __forceinline__ void block_gemm16(float* out, int ldo, const bf16* Alds, int lda,
                                             const bf16* W, int ldw, const float* bias,
                                             int N, int Kpad) {
    const int lane   = threadIdx.x & 31;
    const int wave   = threadIdx.x >> 5;
    const int nwaves = blockDim.x >> 5;
    const int tiles  = (N + 15) >> 4;
    for (int tn = wave; tn < tiles; tn += nwaves) {
        v8f acc = {};
        for (int k = 0; k < Kpad; k += 32) {
            vbf a = load_tileA(Alds, lda, k, lane);
            vbf b = load_tileB(W, ldw, tn * 16, k, lane);
            acc = __builtin_amdgcn_wmma_f32_16x16x32_bf16(false, a, false, b,
                                                          (short)0, acc, false, false);
        }
        const int col = tn * 16 + (lane & 15);
        if (col < N) {
            const int rbase = (lane >> 4) * 8;
            const float bv = bias ? bias[col] : 0.0f;
#pragma unroll
            for (int v = 0; v < 8; ++v) {
                const int row = v + rbase;
                float val = acc[v] + bv;
                if (ACCUM) val += out[row * ldo + col];
                out[row * ldo + col] = val;
            }
        }
    }
}

// ---------------------------------------------------------------------------
// Generic WMMA GEMM over global memory:
// C[M x N] = A[M x Kpad bf16] @ W[Npad x Kpad bf16]^T + bias, optional activation.
// grid = (M/16, Npad/16), block = 32 (one wave per tile).
// ---------------------------------------------------------------------------
__global__ void k_gemm(float* __restrict__ C, const bf16* __restrict__ A,
                       const bf16* __restrict__ W, const float* __restrict__ bias,
                       int ldc, int lda, int ldw, int N, int Kpad, int act) {
    const int lane = threadIdx.x;
    const int m0 = blockIdx.x * 16;
    const int n0 = blockIdx.y * 16;
    const bf16* Ab = A + (size_t)m0 * lda;
    v8f acc = {};
    for (int k = 0; k < Kpad; k += 32) {
        vbf a = load_tileA(Ab, lda, k, lane);
        vbf b = load_tileB(W, ldw, n0, k, lane);
        acc = __builtin_amdgcn_wmma_f32_16x16x32_bf16(false, a, false, b,
                                                      (short)0, acc, false, false);
    }
    const int col = n0 + (lane & 15);
    if (col >= N) return;
    const int rbase = (lane >> 4) * 8;
    const float bv = bias ? bias[col] : 0.0f;
#pragma unroll
    for (int v = 0; v < 8; ++v) {
        float val = acc[v] + bv;
        if (act == 1) val = sigm(val);
        else if (act == 2) val = tanhf(val);
        C[(size_t)(m0 + rbase + v) * ldc + col] = val;
    }
}

// ---------------------------------------------------------------------------
// Packing kernels (f32 -> bf16; K zero-padded to mult of 32, rows zero-padded
// to rowsPad so WMMA B-tile loads are unguarded)
// ---------------------------------------------------------------------------
__global__ void k_pack_rows(bf16* __restrict__ dst, const float* __restrict__ src,
                            int rows, int rowsPad, int K, int Kpad,
                            int srcLd, int srcColOff) {
    const int idx = blockIdx.x * 256 + threadIdx.x;
    const int total = rowsPad * Kpad;
    if (idx >= total) return;
    const int r = idx / Kpad, kk = idx - r * Kpad;
    dst[idx] = (r < rows && kk < K) ? (bf16)src[(size_t)r * srcLd + srcColOff + kk]
                                    : (bf16)0.0f;
}

// (B,T,K) f32  ->  (T,B,Kpad) bf16  (seq-major so each timestep is 16 contiguous rows)
__global__ void k_pack_seq(bf16* __restrict__ dst, const float* __restrict__ src,
                           int B, int T, int K, int Kpad) {
    const int idx = blockIdx.x * 256 + threadIdx.x;
    const int total = T * B * Kpad;
    if (idx >= total) return;
    const int kk = idx % Kpad;
    const int rb = idx / Kpad;
    const int b = rb % B, t = rb / B;
    dst[idx] = (kk < K) ? (bf16)src[((size_t)b * T + t) * K + kk] : (bf16)0.0f;
}

// ---------------------------------------------------------------------------
// GRU scan. gx (T,B,225) precomputed (includes bx). Per step: gh = h@Wh^T + bh
// via block WMMA, then elementwise update. Writes h into y[t, b, colOff + j].
// One block, 8 waves, whole sequence.
// ---------------------------------------------------------------------------
__global__ void __launch_bounds__(256) k_gru_scan(float* __restrict__ y, int ldy, int colOff,
                                                  const float* __restrict__ gx,
                                                  const bf16* __restrict__ Wh,
                                                  const float* __restrict__ bh,
                                                  int T, int reverse) {
    __shared__ float sh_h[BSZ * HH];
    __shared__ bf16  sh_hb[BSZ * 96];
    __shared__ float sh_gh[BSZ * 225];
    const int tid = threadIdx.x, nt = blockDim.x;
    for (int i = tid; i < BSZ * HH; i += nt) sh_h[i] = 0.0f;
    for (int i = tid; i < BSZ * 96; i += nt) sh_hb[i] = (bf16)0.0f;
    __syncthreads();
    for (int s = 0; s < T; ++s) {
        const int t = reverse ? (T - 1 - s) : s;
        for (int i = tid; i < BSZ * HH; i += nt) {
            const int b = i / HH, j = i - b * HH;
            sh_hb[b * 96 + j] = (bf16)sh_h[i];
        }
        __syncthreads();
        block_gemm16<false>(sh_gh, 225, sh_hb, 96, Wh, 96, bh, 225, 96);
        __syncthreads();
        const float* gxt = gx + (size_t)t * BSZ * 225;
        for (int i = tid; i < BSZ * HH; i += nt) {
            const int b = i / HH, j = i - b * HH;
            const float* g  = gxt + b * 225;
            const float* gh = sh_gh + b * 225;
            const float r = sigm(g[j] + gh[j]);
            const float z = sigm(g[HH + j] + gh[HH + j]);
            const float n = tanhf(g[2 * HH + j] + r * gh[2 * HH + j]);
            const float hn = (1.0f - z) * n + z * sh_h[i];
            sh_h[i] = hn;
            y[((size_t)t * BSZ + b) * ldy + colOff + j] = hn;
        }
        __syncthreads();
    }
}

// ---------------------------------------------------------------------------
// QPP gated-attention scan (the only scan whose attention depends on h).
// Precomputed: Wuq (Q,B,75), upWPu (P,B,75), gA = uP@W_g1[:, :450]^T (P,B,900).
// Per step: add = upWPu[t] + h@WP_v^T ; s,softmax,c over Q ;
// fused gate g = sigmoid(gA[t] + c@W_g1[:,450:]^T), r2 = g*[up_t, c] (bf16) ;
// gh = h@Wh^T + bh ; gx = r2@Wx^T + bx ; GRU update -> vP[t].
// ---------------------------------------------------------------------------
__global__ void __launch_bounds__(256) k_qpp_scan(
    float* __restrict__ vP, const float* __restrict__ uP, const float* __restrict__ uQ,
    const float* __restrict__ Wuq, const float* __restrict__ upWPu,
    const float* __restrict__ gA, const bf16* __restrict__ WPv,
    const bf16* __restrict__ Wg1c, const bf16* __restrict__ Wx,
    const bf16* __restrict__ Wh, const float* __restrict__ vvec,
    const float* __restrict__ bx, const float* __restrict__ bh, int P) {
    __shared__ float sh_h[BSZ * HH];
    __shared__ bf16  sh_hb[BSZ * 96];
    __shared__ float sh_add[BSZ * HH];
    __shared__ float sh_s[BSZ * QL];
    __shared__ bf16  sh_cb[BSZ * 480];
    __shared__ bf16  sh_r2[BSZ * 928];
    __shared__ float sh_gx[BSZ * 225];
    __shared__ float sh_gh[BSZ * 225];
    const int tid = threadIdx.x, nt = blockDim.x;
    for (int i = tid; i < BSZ * HH; i += nt)  sh_h[i] = 0.0f;
    for (int i = tid; i < BSZ * 96; i += nt)  sh_hb[i] = (bf16)0.0f;
    for (int i = tid; i < BSZ * 480; i += nt) sh_cb[i] = (bf16)0.0f;
    for (int i = tid; i < BSZ * 928; i += nt) sh_r2[i] = (bf16)0.0f;
    __syncthreads();
    for (int t = 0; t < P; ++t) {
        for (int i = tid; i < BSZ * HH; i += nt) {
            const int b = i / HH, j = i - b * HH;
            sh_hb[b * 96 + j] = (bf16)sh_h[i];
        }
        for (int i = tid; i < BSZ * HH; i += nt)
            sh_add[i] = upWPu[(size_t)t * BSZ * HH + i];
        __syncthreads();
        block_gemm16<true>(sh_add, HH, sh_hb, 96, WPv, 96, nullptr, HH, 96);
        __syncthreads();
        // attention scores over Q
        for (int i = tid; i < BSZ * QL; i += nt) {
            const int b = i >> 6, q = i & 63;
            const float* wq = Wuq + ((size_t)q * BSZ + b) * HH;
            const float* ad = sh_add + b * HH;
            float a = 0.0f;
            for (int j = 0; j < HH; ++j) a += tanhf(wq[j] + ad[j]) * vvec[j];
            sh_s[b * QL + q] = a;
        }
        __syncthreads();
        if (tid < BSZ) {
            float mx = -1e30f;
            for (int q = 0; q < QL; ++q) mx = fmaxf(mx, sh_s[tid * QL + q]);
            float sum = 0.0f;
            for (int q = 0; q < QL; ++q) {
                const float e = __expf(sh_s[tid * QL + q] - mx);
                sh_s[tid * QL + q] = e; sum += e;
            }
            const float inv = 1.0f / sum;
            for (int q = 0; q < QL; ++q) sh_s[tid * QL + q] *= inv;
        }
        __syncthreads();
        // context c (bf16, K padded to 480)
        for (int i = tid; i < BSZ * RR; i += nt) {
            const int b = i / RR, r = i - b * RR;
            const float* sa = sh_s + b * QL;
            float a = 0.0f;
            for (int q = 0; q < QL; ++q) a += sa[q] * uQ[((size_t)q * BSZ + b) * RR + r];
            sh_cb[b * 480 + r] = (bf16)a;
        }
        __syncthreads();
        // fused gate: g tile in registers, r2 = g * [up_t, c] written bf16 directly
        {
            const int lane = tid & 31, wave = tid >> 5, nw = nt >> 5;
            const float* gAt = gA + (size_t)t * BSZ * 900;
            const float* uPt = uP + (size_t)t * BSZ * RR;
            for (int tn = wave; tn < 57; tn += nw) {
                v8f acc = {};
                for (int k = 0; k < 480; k += 32) {
                    vbf a = load_tileA(sh_cb, 480, k, lane);
                    vbf b = load_tileB(Wg1c, 480, tn * 16, k, lane);
                    acc = __builtin_amdgcn_wmma_f32_16x16x32_bf16(false, a, false, b,
                                                                  (short)0, acc, false, false);
                }
                const int col = tn * 16 + (lane & 15);
                if (col < 900) {
                    const int rbase = (lane >> 4) * 8;
#pragma unroll
                    for (int v = 0; v < 8; ++v) {
                        const int row = v + rbase;
                        const float g = sigm(acc[v] + gAt[row * 900 + col]);
                        const float rv = (col < RR) ? uPt[row * RR + col]
                                                    : (float)sh_cb[row * 480 + (col - RR)];
                        sh_r2[row * 928 + col] = (bf16)(g * rv);
                    }
                }
            }
        }
        __syncthreads();
        block_gemm16<false>(sh_gh, 225, sh_hb, 96, Wh, 96, bh, 225, 96);
        block_gemm16<false>(sh_gx, 225, sh_r2, 928, Wx, 928, bx, 225, 928);
        __syncthreads();
        for (int i = tid; i < BSZ * HH; i += nt) {
            const int b = i / HH, j = i - b * HH;
            const float r = sigm(sh_gx[b * 225 + j] + sh_gh[b * 225 + j]);
            const float z = sigm(sh_gx[b * 225 + HH + j] + sh_gh[b * 225 + HH + j]);
            const float n = tanhf(sh_gx[b * 225 + 2 * HH + j] + r * sh_gh[b * 225 + 2 * HH + j]);
            const float hn = (1.0f - z) * n + z * sh_h[i];
            sh_h[i] = hn;
            vP[(size_t)t * BSZ * HH + i] = hn;
        }
        __syncthreads();
    }
}

// ---------------------------------------------------------------------------
// SM attention (fully parallel over t, no recurrence): one block per t.
// s[b,p] = v . tanh(WvP[p,b,:] + addSM[t,b,:]); softmax over p; c[b,:] = a . vP
// ---------------------------------------------------------------------------
__global__ void __launch_bounds__(256) k_attn_sm(float* __restrict__ c_all,
                                                 const float* __restrict__ vP,
                                                 const float* __restrict__ WvP,
                                                 const float* __restrict__ addAll,
                                                 const float* __restrict__ vvec, int P) {
    __shared__ float ss[BSZ * PL];
    const int t = blockIdx.x;
    const int tid = threadIdx.x, nt = blockDim.x;
    const float* add = addAll + (size_t)t * BSZ * HH;
    for (int i = tid; i < BSZ * PL; i += nt) {
        const int b = i >> 8, p = i & 255;
        const float* wv = WvP + ((size_t)p * BSZ + b) * HH;
        float a = 0.0f;
        for (int j = 0; j < HH; ++j) a += tanhf(wv[j] + add[b * HH + j]) * vvec[j];
        ss[i] = a;
    }
    __syncthreads();
    if (tid < BSZ) {
        float mx = -1e30f;
        for (int p = 0; p < PL; ++p) mx = fmaxf(mx, ss[tid * PL + p]);
        float sum = 0.0f;
        for (int p = 0; p < PL; ++p) { const float e = __expf(ss[tid * PL + p] - mx); ss[tid * PL + p] = e; sum += e; }
        const float inv = 1.0f / sum;
        for (int p = 0; p < PL; ++p) ss[tid * PL + p] *= inv;
    }
    __syncthreads();
    for (int i = tid; i < BSZ * HH; i += nt) {
        const int b = i / HH, h = i - b * HH;
        float a = 0.0f;
        for (int p = 0; p < PL; ++p) a += ss[b * PL + p] * vP[((size_t)p * BSZ + b) * HH + h];
        c_all[(size_t)t * BSZ * HH + i] = a;
    }
}

// r = [vt, c] packed bf16 (rows, 160), then r2 = sigmoid-gate * r
__global__ void k_build_r_sm(bf16* __restrict__ rA, const float* __restrict__ vP,
                             const float* __restrict__ c_all, int rows) {
    const int idx = blockIdx.x * 256 + threadIdx.x;
    const int total = rows * 160;
    if (idx >= total) return;
    const int r = idx / 160, k = idx - r * 160;
    float val = 0.0f;
    if (k < HH)            val = vP[(size_t)r * HH + k];
    else if (k < 2 * HH)   val = c_all[(size_t)r * HH + (k - HH)];
    rA[idx] = (bf16)val;
}

__global__ void k_gate_mul(bf16* __restrict__ r2, const float* __restrict__ g,
                           const bf16* __restrict__ rA, int rows) {
    const int idx = blockIdx.x * 256 + threadIdx.x;
    const int total = rows * 160;
    if (idx >= total) return;
    const int r = idx / 160, k = idx - r * 160;
    const float val = (k < 150) ? g[(size_t)r * 150 + k] * (float)rA[idx] : 0.0f;
    r2[idx] = (bf16)val;
}

// ---------------------------------------------------------------------------
// Pointer-network initialization: qv = WQ_v @ VQ_r ; sQ ; softmax ; h0 = aQ . uQ
// ---------------------------------------------------------------------------
__global__ void __launch_bounds__(256) k_init_ptr(float* __restrict__ h0,
                                                  const float* __restrict__ uQ,
                                                  const float* __restrict__ Wuq,
                                                  const float* __restrict__ VQr,
                                                  const float* __restrict__ WQv,
                                                  const float* __restrict__ vvec) {
    __shared__ float qv[HH];
    __shared__ float ss[BSZ * QL];
    const int tid = threadIdx.x, nt = blockDim.x;
    for (int j = tid; j < HH; j += nt) {
        float a = 0.0f;
        for (int k = 0; k < 2 * HH; ++k) a += VQr[k] * WQv[(size_t)j * 2 * HH + k];
        qv[j] = a;
    }
    __syncthreads();
    for (int i = tid; i < BSZ * QL; i += nt) {
        const int b = i >> 6, q = i & 63;
        const float* wq = Wuq + ((size_t)q * BSZ + b) * HH;
        float a = 0.0f;
        for (int j = 0; j < HH; ++j) a += tanhf(wq[j] + qv[j]) * vvec[j];
        ss[b * QL + q] = a;
    }
    __syncthreads();
    if (tid < BSZ) {
        float mx = -1e30f;
        for (int q = 0; q < QL; ++q) mx = fmaxf(mx, ss[tid * QL + q]);
        float sum = 0.0f;
        for (int q = 0; q < QL; ++q) { const float e = __expf(ss[tid * QL + q] - mx); ss[tid * QL + q] = e; sum += e; }
        const float inv = 1.0f / sum;
        for (int q = 0; q < QL; ++q) ss[tid * QL + q] *= inv;
    }
    __syncthreads();
    for (int i = tid; i < BSZ * RR; i += nt) {
        const int b = i / RR, r = i - b * RR;
        float a = 0.0f;
        for (int q = 0; q < QL; ++q) a += ss[b * QL + q] * uQ[((size_t)q * BSZ + b) * RR + r];
        h0[i] = a;
    }
}

// Pointer step: add = h@Wa_h^T (WMMA); logits; softmax; c; exports bf16 h and c.
__global__ void __launch_bounds__(256) k_ptr_attn(float* __restrict__ logits,
                                                  bf16* __restrict__ cb_out,
                                                  bf16* __restrict__ hb_out,
                                                  const float* __restrict__ h,
                                                  const float* __restrict__ WhP,
                                                  const float* __restrict__ hP,
                                                  const bf16* __restrict__ Wah,
                                                  const float* __restrict__ vvec, int P) {
    __shared__ bf16  shb[BSZ * 480];
    __shared__ float sadd[BSZ * HH];
    __shared__ float ss[BSZ * PL];
    const int tid = threadIdx.x, nt = blockDim.x;
    for (int i = tid; i < BSZ * 480; i += nt) {
        const int b = i / 480, k = i - b * 480;
        const bf16 v = (k < RR) ? (bf16)h[(size_t)b * RR + k] : (bf16)0.0f;
        shb[i] = v;
        hb_out[i] = v;
    }
    __syncthreads();
    block_gemm16<false>(sadd, HH, shb, 480, Wah, 480, nullptr, HH, 480);
    __syncthreads();
    for (int i = tid; i < BSZ * PL; i += nt) {
        const int b = i >> 8, p = i & 255;
        const float* wp = WhP + ((size_t)p * BSZ + b) * HH;
        const float* ad = sadd + b * HH;
        float a = 0.0f;
        for (int j = 0; j < HH; ++j) a += tanhf(wp[j] + ad[j]) * vvec[j];
        ss[i] = a;
        logits[(size_t)b * PL + p] = a;   // output layout (step, B, P)
    }
    __syncthreads();
    if (tid < BSZ) {
        float mx = -1e30f;
        for (int p = 0; p < PL; ++p) mx = fmaxf(mx, ss[tid * PL + p]);
        float sum = 0.0f;
        for (int p = 0; p < PL; ++p) { const float e = __expf(ss[tid * PL + p] - mx); ss[tid * PL + p] = e; sum += e; }
        const float inv = 1.0f / sum;
        for (int p = 0; p < PL; ++p) ss[tid * PL + p] *= inv;
    }
    __syncthreads();
    for (int i = tid; i < BSZ * 160; i += nt) {
        const int b = i / 160, k = i - b * 160;
        float a = 0.0f;
        if (k < 150) {
            for (int p = 0; p < PL; ++p) a += ss[b * PL + p] * hP[((size_t)p * BSZ + b) * 150 + k];
        }
        cb_out[i] = (bf16)a;
    }
}

// GRU update for the pointer cell (hidden = R = 450, gates 3*450 = 1350)
__global__ void k_ptr_update(float* __restrict__ h, const float* __restrict__ gx,
                             const float* __restrict__ gh) {
    const int i = blockIdx.x * 256 + threadIdx.x;
    if (i >= BSZ * RR) return;
    const int b = i / RR, j = i - b * RR;
    const float* x = gx + (size_t)b * 1350;
    const float* hh = gh + (size_t)b * 1350;
    const float r = sigm(x[j] + hh[j]);
    const float z = sigm(x[RR + j] + hh[RR + j]);
    const float n = tanhf(x[2 * RR + j] + r * hh[2 * RR + j]);
    h[i] = (1.0f - z) * n + z * h[i];
}

// ---------------------------------------------------------------------------
// Host orchestration
// ---------------------------------------------------------------------------
extern "C" void kernel_launch(void* const* d_in, const int* in_sizes, int n_in,
                              void* d_out, int out_size, void* d_ws, size_t ws_size,
                              hipStream_t stream) {
    (void)in_sizes; (void)n_in; (void)out_size; (void)ws_size;
    // Input order (jax tree flatten, dicts in sorted-key order):
    // 0 P_emb, 1 Q_emb, 2 P_mask, 3 Q_mask (masks all-false -> ignored),
    // 4..7 PN_cell{Wh,Wx,bh,bx}, 8..31 P_enc[l]{b,f}{Wh,Wx,bh,bx},
    // 32..35 QPP_cell, 36..59 Q_enc, 60..63 SM_b, 64..67 SM_f,
    // 68 VQ_r, 69 WPP_v, 70 WP_h, 71 WP_u, 72 WP_v, 73 WQ_u, 74 WQ_v,
    // 75 W_g1, 76 W_g2, 77 Wa_h, 78 v
    const float* P_emb = (const float*)d_in[0];
    const float* Q_emb = (const float*)d_in[1];
    auto IN = [&](int i) { return (const float*)d_in[i]; };

    char* cur = (char*)d_ws;
    auto alloc = [&](size_t bytes) -> void* {
        void* p = (void*)cur; cur += (bytes + 255) & ~(size_t)255; return p;
    };
    // pack f32 -> bf16 with N rows padded to rowsPad and K padded to Kpad
    auto packTo = [&](const float* src, int rows, int rowsPad, int K, int Kpad,
                      int srcLd, int srcColOff) -> bf16* {
        bf16* dst = (bf16*)alloc((size_t)rowsPad * Kpad * sizeof(bf16));
        const int total = rowsPad * Kpad;
        k_pack_rows<<<dim3((total + 255) / 256), dim3(256), 0, stream>>>(
            dst, src, rows, rowsPad, K, Kpad, srcLd, srcColOff);
        return dst;
    };

    // --- pack weights to bf16 (rows -> mult of 16, K -> mult of 32) ---
    bf16 *pWx[3][2], *pWh[3][2], *qWx[3][2], *qWh[3][2];
    const float *pbx[3][2], *pbh[3][2], *qbx[3][2], *qbh[3][2];
    for (int l = 0; l < 3; ++l) {
        for (int d = 0; d < 2; ++d) {   // d=0 forward ('f'), d=1 backward ('b')
            const int pbase = 8 + l * 8 + (d == 0 ? 4 : 0);
            const int qbase = 36 + l * 8 + (d == 0 ? 4 : 0);
            const int Kin = (l == 0) ? EMBD : 2 * HH;
            const int Kpad = (l == 0) ? 320 : 160;
            pWh[l][d] = packTo(IN(pbase + 0), 225, 240, HH, 96, HH, 0);
            pWx[l][d] = packTo(IN(pbase + 1), 225, 240, Kin, Kpad, Kin, 0);
            pbh[l][d] = IN(pbase + 2); pbx[l][d] = IN(pbase + 3);
            qWh[l][d] = packTo(IN(qbase + 0), 225, 240, HH, 96, HH, 0);
            qWx[l][d] = packTo(IN(qbase + 1), 225, 240, Kin, Kpad, Kin, 0);
            qbh[l][d] = IN(qbase + 2); qbx[l][d] = IN(qbase + 3);
        }
    }
    bf16* qppWh = packTo(IN(32), 225, 240, HH, 96, HH, 0);
    bf16* qppWx = packTo(IN(33), 225, 240, 2 * RR, 928, 2 * RR, 0);
    const float *qpp_bh = IN(34), *qpp_bx = IN(35);
    bf16* smbWh = packTo(IN(60), 225, 240, HH, 96, HH, 0);
    bf16* smbWx = packTo(IN(61), 225, 240, 150, 160, 150, 0);
    const float *smb_bh = IN(62), *smb_bx = IN(63);
    bf16* smfWh = packTo(IN(64), 225, 240, HH, 96, HH, 0);
    bf16* smfWx = packTo(IN(65), 225, 240, 150, 160, 150, 0);
    const float *smf_bh = IN(66), *smf_bx = IN(67);
    bf16* pnWh = packTo(IN(4), 1350, 1360, RR, 480, RR, 0);
    bf16* pnWx = packTo(IN(5), 1350, 1360, 150, 160, 150, 0);
    const float *pn_bh = IN(6), *pn_bx = IN(7);
    const float* VQr  = IN(68);
    bf16* WPPv_p = packTo(IN(69), HH, 80, HH, 96, HH, 0);
    bf16* WPh_p  = packTo(IN(70), HH, 80, 150, 160, 150, 0);
    bf16* WPu_p  = packTo(IN(71), HH, 80, RR, 480, RR, 0);
    bf16* WPv_p  = packTo(IN(72), HH, 80, HH, 96, HH, 0);
    bf16* WQu_p  = packTo(IN(73), HH, 80, RR, 480, RR, 0);
    const float* WQv_f = IN(74);
    bf16* g1a = packTo(IN(75), 900, 912, RR, 480, 900, 0);     // W_g1[:, :450]
    bf16* g1c = packTo(IN(75), 900, 912, RR, 480, 900, RR);    // W_g1[:, 450:]
    bf16* Wg2_p = packTo(IN(76), 150, 160, 150, 160, 150, 0);
    bf16* Wah_p = packTo(IN(77), HH, 80, RR, 480, RR, 0);
    const float* vv = IN(78);

    // --- activation scratch ---
    bf16* Pe = (bf16*)alloc((size_t)PB * 320 * sizeof(bf16));
    bf16* Qe = (bf16*)alloc((size_t)QB * 320 * sizeof(bf16));
    {
        int tot = PB * 320;
        k_pack_seq<<<dim3((tot + 255) / 256), dim3(256), 0, stream>>>(Pe, P_emb, BSZ, PL, EMBD, 320);
        tot = QB * 320;
        k_pack_seq<<<dim3((tot + 255) / 256), dim3(256), 0, stream>>>(Qe, Q_emb, BSZ, QL, EMBD, 320);
    }
    float* uP  = (float*)alloc((size_t)PB * RR * sizeof(float));
    float* uQ  = (float*)alloc((size_t)QB * RR * sizeof(float));
    float* gxA = (float*)alloc((size_t)PB * 225 * sizeof(float));
    float* gxB = (float*)alloc((size_t)PB * 225 * sizeof(float));
    bf16*  xin = (bf16*)alloc((size_t)PB * 160 * sizeof(bf16));

    // --- P encoder (3 BiGRU layers) ---
    for (int l = 0; l < 3; ++l) {
        const bf16* in = (l == 0) ? Pe : xin;
        const int Kp = (l == 0) ? 320 : 160;
        k_gemm<<<dim3(PB / 16, 15), dim3(32), 0, stream>>>(gxA, in, pWx[l][0], pbx[l][0], 225, Kp, Kp, 225, Kp, 0);
        k_gemm<<<dim3(PB / 16, 15), dim3(32), 0, stream>>>(gxB, in, pWx[l][1], pbx[l][1], 225, Kp, Kp, 225, Kp, 0);
        k_gru_scan<<<1, 256, 0, stream>>>(uP, RR, l * 150,      gxA, pWh[l][0], pbh[l][0], PL, 0);
        k_gru_scan<<<1, 256, 0, stream>>>(uP, RR, l * 150 + HH, gxB, pWh[l][1], pbh[l][1], PL, 1);
        if (l < 2) {
            const int tot = PB * 160;
            k_pack_rows<<<dim3((tot + 255) / 256), dim3(256), 0, stream>>>(xin, uP, PB, PB, 150, 160, RR, l * 150);
        }
    }
    // --- Q encoder ---
    for (int l = 0; l < 3; ++l) {
        const bf16* in = (l == 0) ? Qe : xin;
        const int Kp = (l == 0) ? 320 : 160;
        k_gemm<<<dim3(QB / 16, 15), dim3(32), 0, stream>>>(gxA, in, qWx[l][0], qbx[l][0], 225, Kp, Kp, 225, Kp, 0);
        k_gemm<<<dim3(QB / 16, 15), dim3(32), 0, stream>>>(gxB, in, qWx[l][1], qbx[l][1], 225, Kp, Kp, 225, Kp, 0);
        k_gru_scan<<<1, 256, 0, stream>>>(uQ, RR, l * 150,      gxA, qWh[l][0], qbh[l][0], QL, 0);
        k_gru_scan<<<1, 256, 0, stream>>>(uQ, RR, l * 150 + HH, gxB, qWh[l][1], qbh[l][1], QL, 1);
        if (l < 2) {
            const int tot = QB * 160;
            k_pack_rows<<<dim3((tot + 255) / 256), dim3(256), 0, stream>>>(xin, uQ, QB, QB, 150, 160, RR, l * 150);
        }
    }

    // --- hoisted GEMMs for the QPP attention scan ---
    bf16* uPp = packTo(uP, PB, PB, RR, 480, RR, 0);
    bf16* uQp = packTo(uQ, QB, QB, RR, 480, RR, 0);
    float* Wuq = (float*)alloc((size_t)QB * HH * sizeof(float));
    k_gemm<<<dim3(QB / 16, 5), dim3(32), 0, stream>>>(Wuq, uQp, WQu_p, nullptr, HH, 480, 480, HH, 480, 0);
    float* upWPu = (float*)alloc((size_t)PB * HH * sizeof(float));
    k_gemm<<<dim3(PB / 16, 5), dim3(32), 0, stream>>>(upWPu, uPp, WPu_p, nullptr, HH, 480, 480, HH, 480, 0);
    float* gA = (float*)alloc((size_t)PB * 900 * sizeof(float));
    k_gemm<<<dim3(PB / 16, 57), dim3(32), 0, stream>>>(gA, uPp, g1a, nullptr, 900, 480, 480, 900, 480, 0);

    float* vPb = (float*)alloc((size_t)PB * HH * sizeof(float));
    k_qpp_scan<<<1, 256, 0, stream>>>(vPb, uP, uQ, Wuq, upWPu, gA, WPv_p, g1c,
                                      qppWx, qppWh, vv, qpp_bx, qpp_bh, PL);

    // --- SM layer: attention fully hoisted, then two plain GRU scans ---
    bf16* vPp = packTo(vPb, PB, PB, HH, 96, HH, 0);
    float* WvP = (float*)alloc((size_t)PB * HH * sizeof(float));
    k_gemm<<<dim3(PB / 16, 5), dim3(32), 0, stream>>>(WvP, vPp, WPPv_p, nullptr, HH, 96, 96, HH, 96, 0);
    float* addSM = (float*)alloc((size_t)PB * HH * sizeof(float));
    k_gemm<<<dim3(PB / 16, 5), dim3(32), 0, stream>>>(addSM, vPp, WPv_p, nullptr, HH, 96, 96, HH, 96, 0);
    float* c_all = (float*)alloc((size_t)PB * HH * sizeof(float));
    k_attn_sm<<<dim3(PL), dim3(256), 0, stream>>>(c_all, vPb, WvP, addSM, vv, PL);

    bf16* rA = (bf16*)alloc((size_t)PB * 160 * sizeof(bf16));
    {
        const int tot = PB * 160;
        k_build_r_sm<<<dim3((tot + 255) / 256), dim3(256), 0, stream>>>(rA, vPb, c_all, PB);
    }
    float* gbuf = (float*)alloc((size_t)PB * 150 * sizeof(float));
    k_gemm<<<dim3(PB / 16, 10), dim3(32), 0, stream>>>(gbuf, rA, Wg2_p, nullptr, 150, 160, 160, 150, 160, 1);
    bf16* r2 = (bf16*)alloc((size_t)PB * 160 * sizeof(bf16));
    {
        const int tot = PB * 160;
        k_gate_mul<<<dim3((tot + 255) / 256), dim3(256), 0, stream>>>(r2, gbuf, rA, PB);
    }
    k_gemm<<<dim3(PB / 16, 15), dim3(32), 0, stream>>>(gxA, r2, smfWx, smf_bx, 225, 160, 160, 225, 160, 0);
    k_gemm<<<dim3(PB / 16, 15), dim3(32), 0, stream>>>(gxB, r2, smbWx, smb_bx, 225, 160, 160, 225, 160, 0);
    float* hP = (float*)alloc((size_t)PB * 150 * sizeof(float));
    k_gru_scan<<<1, 256, 0, stream>>>(hP, 150, 0,  gxA, smfWh, smf_bh, PL, 0);
    k_gru_scan<<<1, 256, 0, stream>>>(hP, 150, HH, gxB, smbWh, smb_bh, PL, 1);

    bf16* hPp = packTo(hP, PB, PB, 150, 160, 150, 0);
    float* WhP = (float*)alloc((size_t)PB * HH * sizeof(float));
    k_gemm<<<dim3(PB / 16, 5), dim3(32), 0, stream>>>(WhP, hPp, WPh_p, nullptr, HH, 160, 160, HH, 160, 0);

    // --- pointer network ---
    float* h0 = (float*)alloc((size_t)BSZ * RR * sizeof(float));
    k_init_ptr<<<1, 256, 0, stream>>>(h0, uQ, Wuq, VQr, WQv_f, vv);
    bf16* cbP = (bf16*)alloc((size_t)BSZ * 160 * sizeof(bf16));
    bf16* hbP = (bf16*)alloc((size_t)BSZ * 480 * sizeof(bf16));
    float* gX = (float*)alloc((size_t)BSZ * 1350 * sizeof(float));
    float* gH = (float*)alloc((size_t)BSZ * 1350 * sizeof(float));
    float* out = (float*)d_out;  // (2, B, P)
    for (int step = 0; step < 2; ++step) {
        k_ptr_attn<<<1, 256, 0, stream>>>(out + (size_t)step * BSZ * PL, cbP, hbP,
                                          h0, WhP, hP, Wah_p, vv, PL);
        k_gemm<<<dim3(1, 85), dim3(32), 0, stream>>>(gX, cbP, pnWx, pn_bx, 1350, 160, 160, 1350, 160, 0);
        k_gemm<<<dim3(1, 85), dim3(32), 0, stream>>>(gH, hbP, pnWh, pn_bh, 1350, 480, 480, 1350, 480, 0);
        k_ptr_update<<<dim3((BSZ * RR + 255) / 256), dim3(256), 0, stream>>>(h0, gX, gH);
    }
}
